// Attention_2_78323023610197
// MI455X (gfx1250) — matrix operand order repaired and checked by reading, not compiled, not run on hardware
//
#include <hip/hip_runtime.h>

typedef __attribute__((ext_vector_type(16))) _Float16 v16h;
typedef __attribute__((ext_vector_type(8)))  float    v8f;

// Per-lane j index set (A/B fragments use K = e+16*half instead) of
// the transposed-D per-lane row gather (j = r + 8*half + 16*tile).
//   jmap(e,h) = (e<8 ? e : e+8) + 8*h
__device__ __forceinline__ int jmap(int e, int half) {
    return ((e < 8) ? e : (e + 8)) + 8 * half;
}

__global__ __launch_bounds__(32) void attn_fused(
    const float* __restrict__ sd,    // [B,32]  source_distance
    const float* __restrict__ ctx,   // [B,32,192] context
    const float* __restrict__ kern,  // [4,32,32] kernels
    const float* __restrict__ bias,  // [4,32] biases
    const float* __restrict__ gW,    // [32,32] gate_W
    const float* __restrict__ gb,    // [32]   gate_b
    const float* __restrict__ gwh,   // [4]    gate_weights
    const float* __restrict__ gbh,   // [4]    gate_bias
    float* __restrict__ out)         // [B,192]
{
    __shared__ float wlds[512];          // final combined weights [16][32]

    const int lane = threadIdx.x;        // 0..31 (wave32)
    const int half = lane >> 4;          // 0 or 1
    const int l16  = lane & 15;
    const int b0   = blockIdx.x * 16;    // 16 batch rows per block

    // ---- head-gate softmax over H=4 (uniform per lane) ----
    float hg[4];
    {
        float t[4], m = -1e30f, s = 0.f;
        #pragma unroll
        for (int h = 0; h < 4; ++h) { t[h] = gwh[h] + gbh[h]; m = fmaxf(m, t[h]); }
        #pragma unroll
        for (int h = 0; h < 4; ++h) { t[h] = __expf(t[h] - m); s += t[h]; }
        float inv = 1.0f / s;
        #pragma unroll
        for (int h = 0; h < 4; ++h) hg[h] = t[h] * inv;
    }

    // ---- B fragment: simi^T (32x16): element e -> K = e+16*half, N = l16 ----
    // 16 contiguous floats per lane -> b128 loads; no LDS staging.
    v16h bsimi;
    {
        const float* sdrow = sd + (size_t)(b0 + l16) * 32 + 16 * half;
        #pragma unroll
        for (int e = 0; e < 16; ++e) {
            float v = sdrow[e];
            bsimi[e] = (_Float16)__expf(-0.5f * v * v);
        }
    }

    // ---- A fragments: gate_W^T tiles (constant across heads) ----
    // agw_t[e] = gate_W^T[j2 = l16+16t][j = e+16*half] = gW[j*32 + l16 + 16t]
    v16h agw0, agw1;
    #pragma unroll
    for (int e = 0; e < 16; ++e) {
        int j = e + 16 * half;
        agw0[e] = (_Float16)gW[j * 32 + l16];
        agw1[e] = (_Float16)gW[j * 32 + l16 + 16];
    }
    // gate bias in transposed-D per-lane layout (constant across heads)
    float gbv[16];
    #pragma unroll
    for (int e = 0; e < 16; ++e) gbv[e] = gb[jmap(e, half)];

    float wc[16];                        // combined weights, per-lane row b=l16
    #pragma unroll
    for (int e = 0; e < 16; ++e) wc[e] = 0.f;

    for (int h = 0; h < 4; ++h) {
        // A fragments: kernels[h]^T tiles; ak_t[e] = kern[h][k=e+16*half][l16+16t]
        v16h ak0, ak1;
        #pragma unroll
        for (int e = 0; e < 16; ++e) {
            int j = e + 16 * half;
            ak0[e] = (_Float16)kern[h * 1024 + j * 32 + l16];
            ak1[e] = (_Float16)kern[h * 1024 + j * 32 + l16 + 16];
        }
        float biasv[16];
        #pragma unroll
        for (int e = 0; e < 16; ++e) biasv[e] = bias[h * 32 + jmap(e, half)];

        v8f c0, c1;
        #pragma unroll
        for (int r = 0; r < 8; ++r) { c0[r] = biasv[r]; c1[r] = biasv[r + 8]; }
        // logits^T = kernels[h]^T @ simi^T  (+bias preloaded in C)
        // -> lane L holds logits[b=l16][j = jmap(e,half)] as {c0,c1}
        c0 = __builtin_amdgcn_wmma_f32_16x16x32_f16(false, ak0, false, bsimi, (short)0, c0, false, false);
        c1 = __builtin_amdgcn_wmma_f32_16x16x32_f16(false, ak1, false, bsimi, (short)0, c1, false, false);

        // ---- softmax over row b=l16: 16 regs here + 16 in partner lane (xor 16)
        float lv[16];
        #pragma unroll
        for (int r = 0; r < 8; ++r) { lv[r] = c0[r]; lv[r + 8] = c1[r]; }
        float m = lv[0];
        #pragma unroll
        for (int e = 1; e < 16; ++e) m = fmaxf(m, lv[e]);
        m = fmaxf(m, __shfl_xor(m, 16, 32));
        float w[16], s = 0.f;
        #pragma unroll
        for (int e = 0; e < 16; ++e) { w[e] = __expf(lv[e] - m); s += w[e]; }
        s += __shfl_xor(s, 16, 32);
        float inv = 1.0f / s;
        #pragma unroll
        for (int e = 0; e < 16; ++e) w[e] *= inv;

        // ---- B fragment of weight^T: B[e] = w_row[j = e + 16*half].
        // Lane has the interleaved half of the row; 8-value partner exchange fills it.
        float recvv[8];
        #pragma unroll
        for (int ee = 0; ee < 8; ++ee) {
            float send = half ? w[ee] : w[ee + 8];
            recvv[ee] = __shfl_xor(send, 16, 32);
        }
        v16h bw;
        #pragma unroll
        for (int e = 0; e < 8; ++e) {
            bw[e]     = (_Float16)(half ? recvv[e] : w[e]);
            bw[e + 8] = (_Float16)(half ? w[e + 8] : recvv[e]);
        }

        // gate^T = gate_W^T @ weight^T (+gate_b in C) -> same per-lane layout as w
        v8f g0, g1;
        #pragma unroll
        for (int r = 0; r < 8; ++r) { g0[r] = gbv[r]; g1[r] = gbv[r + 8]; }
        g0 = __builtin_amdgcn_wmma_f32_16x16x32_f16(false, agw0, false, bw, (short)0, g0, false, false);
        g1 = __builtin_amdgcn_wmma_f32_16x16x32_f16(false, agw1, false, bw, (short)0, g1, false, false);

        // gated = w * sigmoid(g); fold head gate; all in registers
        const float hgh = hg[h];
        #pragma unroll
        for (int e = 0; e < 8; ++e) {
            float s0 = 1.0f / (1.0f + __expf(-g0[e]));
            float s1 = 1.0f / (1.0f + __expf(-g1[e]));
            wc[e]     += hgh * w[e]     * s0;
            wc[e + 8] += hgh * w[e + 8] * s1;
        }
    }

    // ---- single LDS write of combined weight tile w[b_local][k] ----
    #pragma unroll
    for (int e = 0; e < 16; ++e)
        wlds[l16 * 32 + jmap(e, half)] = wc[e];
    __syncthreads();

    // ---- Phase B: out[b,:] = w[b,:] @ context[b,:,:]  (pure streaming) ----
    // 192 floats = 96 float2 = exactly 3 x 32 lanes -> fully coalesced 256B loads.
    const float2* ctx2 = (const float2*)(ctx + (size_t)b0 * 32 * 192);
    float2*       out2 = (float2*)(out + (size_t)b0 * 192);

    for (int bi = 0; bi < 16; ++bi) {
        const float2* cb = ctx2 + (size_t)bi * 32 * 96;
        if (bi < 15) {  // speculative prefetch of next row-block (24 KB span)
            const float* nb = (const float*)(ctx2 + (size_t)(bi + 1) * 32 * 96);
            __builtin_prefetch(nb + lane * 192, 0, 0);   // one line per k-row start
        }
        const float* wrow = &wlds[bi * 32];
        float2 a0 = {0.f, 0.f}, a1 = {0.f, 0.f}, a2 = {0.f, 0.f};
        #pragma unroll 8
        for (int k = 0; k < 32; ++k) {
            float wv = wrow[k];                    // LDS broadcast
            float2 v0 = cb[k * 96 + lane];
            float2 v1 = cb[k * 96 + 32 + lane];
            float2 v2 = cb[k * 96 + 64 + lane];
            a0.x += wv * v0.x; a0.y += wv * v0.y;
            a1.x += wv * v1.x; a1.y += wv * v1.y;
            a2.x += wv * v2.x; a2.y += wv * v2.y;
        }
        out2[bi * 96 + lane]      = a0;
        out2[bi * 96 + 32 + lane] = a1;
        out2[bi * 96 + 64 + lane] = a2;
    }
}

extern "C" void kernel_launch(void* const* d_in, const int* in_sizes, int n_in,
                              void* d_out, int out_size, void* d_ws, size_t ws_size,
                              hipStream_t stream) {
    (void)n_in; (void)out_size; (void)d_ws; (void)ws_size;
    const float* sd   = (const float*)d_in[0];
    const float* ctx  = (const float*)d_in[1];
    const float* kern = (const float*)d_in[2];
    const float* bias = (const float*)d_in[3];
    const float* gW   = (const float*)d_in[4];
    const float* gb   = (const float*)d_in[5];
    const float* gwh  = (const float*)d_in[6];
    const float* gbh  = (const float*)d_in[7];
    float* out = (float*)d_out;

    const int B = in_sizes[0] / 32;          // 32768
    dim3 grid(B / 16), block(32);
    hipLaunchKernelGGL(attn_fused, grid, block, 0, stream,
                       sd, ctx, kern, bias, gW, gb, gwh, gbh, out);
}
